// GraphCNN_32358283608240
// MI455X (gfx1250) — compile-verified, matrix operand order checked by reading, not run on hardware
//
#include <hip/hip_runtime.h>

// ---------------------------------------------------------------------------
// CDNA5 (gfx1250) GIN forward: WMMA bf16 GEMMs + streaming graph kernels
// ---------------------------------------------------------------------------

typedef __attribute__((ext_vector_type(16))) __bf16 bf16x16;
typedef __attribute__((ext_vector_type(8)))  float  floatx8;

#define D_H      128
#define D_ATT    640     // 128 * (L+1)
#define D_PI     16
#define D_OUT    10
#define LDSSTR   136     // padded LDS row stride (bank-conflict free frag loads)

// ---------------------------------------------------------------------------
// Fragment loader: 16-bit A-matrix 16x32 layout (ISA 7.12.2).
// Lane L (0..15): row M=L, K = k0+0..7 (vgpr0..3) and k0+16..23 (vgpr4..7)
// Lane L (16..31): row M=L-16, K = k0+8..15 and k0+24..31.
// B uses the same loader on a transposed LDS tile (sB[n][k]).
// ---------------------------------------------------------------------------
__device__ __forceinline__ bf16x16 load_frag(const __bf16* __restrict__ base,
                                             int lane, int k0) {
  const __bf16* p = base + (lane & 15) * LDSSTR + k0 + ((lane >> 4) << 3);
  bf16x16 f;
#pragma unroll
  for (int j = 0; j < 4; ++j) {
    f[2 * j]     = p[2 * j];
    f[2 * j + 1] = p[2 * j + 1];
    f[8 + 2 * j]     = p[16 + 2 * j];
    f[8 + 2 * j + 1] = p[16 + 2 * j + 1];
  }
  return f;
}

// ---------------------------------------------------------------------------
// Z[r][n] = sum_k A[r][k] * W[k][n] + bias[n]   (A strided by lda)
// 128 threads = 4 waves; block tile 64 rows x 128 cols; K=128 in 4 steps.
// ---------------------------------------------------------------------------
__global__ __launch_bounds__(128)
void gemm128_bias(const float* __restrict__ A, int lda,
                  const float* __restrict__ W,
                  const float* __restrict__ bias,
                  float* __restrict__ Z, int nrows) {
  __shared__ __bf16 sA[64 * LDSSTR];
  __shared__ __bf16 sB[128 * LDSSTR];   // transposed: sB[n][k] = W[k][n]

  const int tid = threadIdx.x;
  const int rowBase = blockIdx.x * 64;

  for (int idx = tid; idx < D_H * D_H; idx += 128) {
    int k = idx >> 7, n = idx & 127;
    sB[n * LDSSTR + k] = (__bf16)W[idx];
  }
  for (int idx = tid; idx < 64 * D_H; idx += 128) {
    int r = idx >> 7, c = idx & 127;
    int row = rowBase + r;
    float v = (row < nrows) ? A[(size_t)row * lda + c] : 0.0f;
    sA[r * LDSSTR + c] = (__bf16)v;
  }
  __syncthreads();

  const int wave = tid >> 5;
  const int lane = tid & 31;
  const __bf16* aBase = &sA[(wave * 16) * LDSSTR];

  floatx8 acc[8];
#pragma unroll
  for (int t = 0; t < 8; ++t) {
    float bv = bias[t * 16 + (lane & 15)];
#pragma unroll
    for (int i = 0; i < 8; ++i) acc[t][i] = bv;
  }

#pragma unroll
  for (int kk = 0; kk < D_H; kk += 32) {
    bf16x16 af = load_frag(aBase, lane, kk);
#pragma unroll
    for (int t = 0; t < 8; ++t) {
      bf16x16 bf = load_frag(&sB[(t * 16) * LDSSTR], lane, kk);
      acc[t] = __builtin_amdgcn_wmma_f32_16x16x32_bf16(
          false, af, false, bf, (short)0, acc[t], false, false);
    }
  }

  // C/D layout: vgpr i, lanes 0-15: M=i, N=lane; lanes 16-31: M=8+i, N=lane-16
  const int colL = lane & 15;
  const int rowH = (lane >> 4) << 3;
#pragma unroll
  for (int t = 0; t < 8; ++t) {
#pragma unroll
    for (int i = 0; i < 8; ++i) {
      int r = rowBase + wave * 16 + rowH + i;
      if (r < nrows) Z[(size_t)r * D_H + (t * 16 + colL)] = acc[t][i];
    }
  }
}

// ---------------------------------------------------------------------------
__global__ void zero_f32(float* __restrict__ p, int n) {
  int i = blockIdx.x * blockDim.x + threadIdx.x;
  if (i < n) p[i] = 0.0f;
}

__global__ void copy_x_to_hr(const float* __restrict__ x,
                             float* __restrict__ hr, int n) {
  size_t i = (size_t)blockIdx.x * blockDim.x + threadIdx.x;
  if (i >= (size_t)n * D_H) return;
  size_t r = i >> 7; int c = (int)(i & 127);
  hr[r * D_ATT + c] = x[i];
}

__global__ void init_pooled(const float* __restrict__ h, int ldh,
                            const float* __restrict__ eps, int l,
                            float* __restrict__ pooled, int n) {
  size_t i = (size_t)blockIdx.x * blockDim.x + threadIdx.x;
  if (i >= (size_t)n * D_H) return;
  size_t r = i >> 7; int c = (int)(i & 127);
  pooled[i] = (1.0f + eps[l]) * h[r * (size_t)ldh + c];
}

// one thread per (edge, 4-feature chunk): float4 gather + 4 atomic adds
__global__ void edge_scatter(const float* __restrict__ h, int ldh,
                             const int* __restrict__ esrc,
                             const int* __restrict__ edst,
                             float* __restrict__ pooled, int E) {
  size_t idx = (size_t)blockIdx.x * blockDim.x + threadIdx.x;
  if (idx >= (size_t)E * 32) return;
  int e  = (int)(idx >> 5);
  int c4 = (int)(idx & 31) << 2;
  int s = esrc[e], d = edst[e];
  const float4 v = *(const float4*)(h + (size_t)s * ldh + c4);
  float* p = pooled + (size_t)d * D_H + c4;
  atomicAdd(p + 0, v.x);
  atomicAdd(p + 1, v.y);
  atomicAdd(p + 2, v.z);
  atomicAdd(p + 3, v.w);
}

// per-column sum / sum-of-squares: stats[0..127]=sum, stats[128..255]=sumsq
__global__ __launch_bounds__(256)
void colstats(const float* __restrict__ Z, float* __restrict__ stats, int n) {
  const int col = threadIdx.x & 127;
  const int sub = threadIdx.x >> 7;  // 0 or 1
  float s = 0.0f, sq = 0.0f;
  for (int r = blockIdx.x * 2 + sub; r < n; r += gridDim.x * 2) {
    float v = Z[(size_t)r * D_H + col];
    s += v; sq += v * v;
  }
  __shared__ float sh[256], shq[256];
  sh[threadIdx.x] = s; shq[threadIdx.x] = sq;
  __syncthreads();
  if (sub == 0) {
    s  = sh[col]  + sh[col + 128];
    sq = shq[col] + shq[col + 128];
    atomicAdd(&stats[col], s);
    atomicAdd(&stats[128 + col], sq);
  }
}

__global__ void bn_relu(const float* __restrict__ Z,
                        const float* __restrict__ stats,
                        const float* __restrict__ gamma,
                        const float* __restrict__ beta,
                        float* __restrict__ out, int ldo, int n, float invN) {
  size_t i = (size_t)blockIdx.x * blockDim.x + threadIdx.x;
  if (i >= (size_t)n * D_H) return;
  int c = (int)(i & 127); size_t r = i >> 7;
  float mean = stats[c] * invN;
  float var  = stats[128 + c] * invN - mean * mean;
  float v = (Z[i] - mean) * rsqrtf(var + 1e-5f) * gamma[c] + beta[c];
  out[r * (size_t)ldo + c] = fmaxf(v, 0.0f);
}

// attention readout: one block per node; dot(hr[n], w_att)+b then
// weighted scatter-add into the node's graph embedding.
__global__ __launch_bounds__(128)
void readout(const float* __restrict__ hr, const float* __restrict__ w_att,
             const float* __restrict__ b_att, const int* __restrict__ seg,
             float* __restrict__ gemb, int n) {
  const int node = blockIdx.x;
  if (node >= n) return;
  const int tid = threadIdx.x;
  const float* hp = hr + (size_t)node * D_ATT;
  float part = 0.0f;
  for (int c = tid; c < D_ATT; c += 128) part += hp[c] * w_att[c];
  __shared__ float sh[128];
  sh[tid] = part;
  __syncthreads();
  for (int s = 64; s > 0; s >>= 1) {
    if (tid < s) sh[tid] += sh[tid + s];
    __syncthreads();
  }
  const float a = sh[0] + b_att[0];
  const int g = seg[node];
  float* gp = gemb + (size_t)g * D_ATT;
  for (int c = tid; c < D_ATT; c += 128) atomicAdd(&gp[c], a * hp[c]);
}

// score = concat(gemb, pi) @ W_out + b_out   ([G,656] x [656,10] -> [G,10])
__global__ void final_out(const float* __restrict__ gemb,
                          const float* __restrict__ pi,
                          const float* __restrict__ W_out,
                          const float* __restrict__ b_out,
                          float* __restrict__ out, int G) {
  int idx = blockIdx.x * blockDim.x + threadIdx.x;
  if (idx >= G * D_OUT) return;
  int g = idx / D_OUT, o = idx % D_OUT;
  float acc = b_out[o];
  const float* ge = gemb + (size_t)g * D_ATT;
  for (int k = 0; k < D_ATT; ++k) acc += ge[k] * W_out[k * D_OUT + o];
  const float* pp = pi + (size_t)g * D_PI;
  for (int k = 0; k < D_PI; ++k) acc += pp[k] * W_out[(D_ATT + k) * D_OUT + o];
  out[idx] = acc;
}

// ---------------------------------------------------------------------------
extern "C" void kernel_launch(void* const* d_in, const int* in_sizes, int n_in,
                              void* d_out, int out_size, void* d_ws,
                              size_t ws_size, hipStream_t stream) {
  const float* x     = (const float*)d_in[0];
  const float* pi    = (const float*)d_in[1];
  const int*   esrc  = (const int*)  d_in[2];
  const int*   edst  = (const int*)  d_in[3];
  const int*   seg   = (const int*)  d_in[4];
  const float* eps   = (const float*)d_in[5];
  const float* W1    = (const float*)d_in[6];
  const float* b1    = (const float*)d_in[7];
  const float* g1    = (const float*)d_in[8];
  const float* be1   = (const float*)d_in[9];
  const float* W2    = (const float*)d_in[10];
  const float* b2    = (const float*)d_in[11];
  const float* gbn   = (const float*)d_in[12];
  const float* bbn   = (const float*)d_in[13];
  const float* w_att = (const float*)d_in[14];
  const float* b_att = (const float*)d_in[15];
  const float* W_out = (const float*)d_in[16];
  const float* b_out = (const float*)d_in[17];

  const int N = in_sizes[0] / D_H;
  const int E = in_sizes[2];
  const int G = in_sizes[1] / D_PI;
  const int L = in_sizes[5];

  // workspace layout (floats)
  float* hr     = (float*)d_ws;                    // [N, 640]
  float* pooled = hr + (size_t)N * D_ATT;          // [N, 128]
  float* zbuf   = pooled + (size_t)N * D_H;        // [N, 128]
  float* stats  = zbuf + (size_t)N * D_H;          // [256]
  float* gemb   = stats + 256;                     // [G, 640]

  const float invN = 1.0f / (float)N;
  const int elemGrid = (int)(((size_t)N * D_H + 255) / 256);
  const int gemmGrid = (N + 63) / 64;
  const int edgeGrid = (int)(((size_t)E * 32 + 255) / 256);

  copy_x_to_hr<<<elemGrid, 256, 0, stream>>>(x, hr, N);

  for (int l = 0; l < L; ++l) {
    const float* hin = hr + (size_t)l * D_H;   // row stride 640

    init_pooled<<<elemGrid, 256, 0, stream>>>(hin, D_ATT, eps, l, pooled, N);
    edge_scatter<<<edgeGrid, 256, 0, stream>>>(hin, D_ATT, esrc, edst, pooled, E);

    // z1 = pooled @ W1[l] + b1[l] ; BN+ReLU -> pooled (reuse)
    gemm128_bias<<<gemmGrid, 128, 0, stream>>>(pooled, D_H, W1 + (size_t)l * D_H * D_H,
                                               b1 + (size_t)l * D_H, zbuf, N);
    zero_f32<<<1, 256, 0, stream>>>(stats, 256);
    colstats<<<256, 256, 0, stream>>>(zbuf, stats, N);
    bn_relu<<<elemGrid, 256, 0, stream>>>(zbuf, stats, g1 + (size_t)l * D_H,
                                          be1 + (size_t)l * D_H, pooled, D_H, N, invN);

    // z2 = y @ W2[l] + b2[l] ; BN+ReLU -> hr slice l+1
    gemm128_bias<<<gemmGrid, 128, 0, stream>>>(pooled, D_H, W2 + (size_t)l * D_H * D_H,
                                               b2 + (size_t)l * D_H, zbuf, N);
    zero_f32<<<1, 256, 0, stream>>>(stats, 256);
    colstats<<<256, 256, 0, stream>>>(zbuf, stats, N);
    bn_relu<<<elemGrid, 256, 0, stream>>>(zbuf, stats, gbn + (size_t)l * D_H,
                                          bbn + (size_t)l * D_H,
                                          hr + (size_t)(l + 1) * D_H, D_ATT, N, invN);
  }

  zero_f32<<<(G * D_ATT + 255) / 256, 256, 0, stream>>>(gemb, G * D_ATT);
  readout<<<N, 128, 0, stream>>>(hr, w_att, b_att, seg, gemb, N);
  final_out<<<(G * D_OUT + 255) / 256, 256, 0, stream>>>(gemb, pi, W_out, b_out,
                                                         (float*)d_out, G);
}
